// TRMAttention_21586505630342
// MI455X (gfx1250) — matrix operand order repaired and checked by reading, not compiled, not run on hardware
//
#include <hip/hip_runtime.h>
#include <math.h>

typedef __bf16 bf16;
typedef __attribute__((ext_vector_type(4)))  bf16  v4bf;
typedef __attribute__((ext_vector_type(8)))  bf16  v8bf;
typedef __attribute__((ext_vector_type(16))) bf16  v16bf;
typedef __attribute__((ext_vector_type(4)))  float v4f;
typedef __attribute__((ext_vector_type(8)))  float v8f;
typedef __attribute__((ext_vector_type(4)))  int   v4i;

#define HIDDEN 256
#define NHEAD 4
#define HDIM 64
#define BATCH 16
#define SEQ 2048

// log2(e) / sqrt(HDIM): softmax done in exp2 domain
#define SOFTMAX_SCL (0.125f * 1.44269504088896f)

__device__ __forceinline__ v16bf cat16(v8bf lo, v8bf hi) {
    return __builtin_shufflevector(lo, hi, 0, 1, 2, 3, 4, 5, 6, 7,
                                   8, 9, 10, 11, 12, 13, 14, 15);
}

// ---- CDNA5 async global->LDS copy (ASYNCcnt-tracked) ----------------------
// Signature (from compiler diagnostic): (as1 v4i*, as3 v4i*, imm offset, cpol)
#if __has_builtin(__builtin_amdgcn_global_load_async_to_lds_b128)
#define ASYNC_B128(g, l)                                                     \
    __builtin_amdgcn_global_load_async_to_lds_b128(                          \
        (__attribute__((address_space(1))) v4i*)(g),                         \
        (__attribute__((address_space(3))) v4i*)(l), 0, 0)
#else
#define ASYNC_B128(g, l)                                                     \
    do {                                                                     \
        unsigned long long _ga = (unsigned long long)(g);                    \
        unsigned int _la =                                                   \
            (unsigned int)(size_t)(__attribute__((address_space(3))) void*)(l); \
        asm volatile("global_load_async_to_lds_b128 %0, %1, off"             \
                     :: "v"(_la), "v"(_ga) : "memory");                      \
    } while (0)
#endif

#if __has_builtin(__builtin_amdgcn_s_wait_asynccnt)
#define WAIT_ASYNC() __builtin_amdgcn_s_wait_asynccnt(0)
#else
#define WAIT_ASYNC() asm volatile("s_wait_asynccnt 0" ::: "memory")
#endif

// ---------------------------------------------------------------------------
// Kernel 0: bulk f32 -> bf16 conversion (packed v_cvt hardware path)
// ---------------------------------------------------------------------------
__global__ void cvt_kernel(const float* __restrict__ src, bf16* __restrict__ dst,
                           int n4) {
    int i = blockIdx.x * blockDim.x + threadIdx.x;
    if (i < n4) {
        v4f f = *(const v4f*)(src + (size_t)i * 4);
        v4bf o;
        o[0] = (bf16)f[0]; o[1] = (bf16)f[1]; o[2] = (bf16)f[2]; o[3] = (bf16)f[3];
        *(v4bf*)(dst + (size_t)i * 4) = o;
    }
}

// ---------------------------------------------------------------------------
// Kernel 1: fused QKV projection (bf16 operands, f32 accumulate).
// M = 32768, N = 768, K = 256.  One 16x16 tile per wave, 8 WMMAs.
// Scatter to Q,K [B,H,T,D] bf16 and V^T [B,H,D,T] bf16.
// ---------------------------------------------------------------------------
__global__ void qkv_kernel(const bf16* __restrict__ xb,
                           const bf16* __restrict__ Wqkv,
                           const float* __restrict__ bqkv,
                           bf16* __restrict__ Qb, bf16* __restrict__ Kb,
                           bf16* __restrict__ Vt) {
    const int lane = threadIdx.x & 31;
    const int w    = threadIdx.x >> 5;
    const int h    = lane >> 4;
    const int m16  = lane & 15;
    const int ntile = blockIdx.x;               // 0..47
    const int mtile = blockIdx.y * 4 + w;       // 0..2047
    const long mbase = (long)mtile * 16;
    const int  nbase = ntile * 16;

    v8f c = {};
    #pragma unroll
    for (int kc = 0; kc < 8; ++kc) {
        const bf16* arow = xb + (mbase + m16) * 256 + kc * 32;
        v16bf a = cat16(*(const v8bf*)(arow + h * 8),
                        *(const v8bf*)(arow + 16 + h * 8));
        const bf16* brow = Wqkv + (long)(nbase + m16) * 256 + kc * 32;
        v16bf b = cat16(*(const v8bf*)(brow + h * 8), *(const v8bf*)(brow + 16 + h * 8));
        c = __builtin_amdgcn_wmma_f32_16x16x32_bf16(false, a, false, b,
                                                    (short)0, c, false, false);
    }

    const int n    = nbase + m16;
    const float bias = bqkv[n];
    const int sel  = n >> 8;
    const int c0   = n & 255;
    const int head = c0 >> 6;
    const int d    = c0 & 63;
    #pragma unroll
    for (int i = 0; i < 8; ++i) {
        long m = mbase + 8 * h + i;
        int  bidx = (int)(m >> 11);
        int  t    = (int)(m & 2047);
        bf16 v = (bf16)(c[i] + bias);
        long bh = (long)(bidx * NHEAD + head);
        if (sel == 0)      Qb[(bh * SEQ + t) * HDIM + d] = v;
        else if (sel == 1) Kb[(bh * SEQ + t) * HDIM + d] = v;
        else               Vt[(bh * HDIM + d) * SEQ + t] = v;
    }
}

// ---------------------------------------------------------------------------
// Kernel 2: flash attention.  4 waves/block share one (b,head); K/V tiles are
// staged once per block into LDS via async global->LDS DMA, double-buffered.
// Each wave owns a 16-query tile; online softmax over 64 key-tiles of 32.
// ---------------------------------------------------------------------------
__global__ void attn_kernel(const bf16* __restrict__ Qb,
                            const bf16* __restrict__ Kb,
                            const bf16* __restrict__ Vt,
                            bf16* __restrict__ Y) {
    __shared__ __align__(16) bf16 Klds[2][32 * 64];   // [key][d]   8 KB
    __shared__ __align__(16) bf16 Vlds[2][64 * 32];   // [d][key]   8 KB
    __shared__ __align__(16) bf16 Plds[4][16][32];    // per-wave P 4 KB

    const int lane = threadIdx.x & 31;
    const int w    = threadIdx.x >> 5;
    const int h    = lane >> 4;
    const int nn   = lane & 15;
    const int b    = blockIdx.z;
    const int head = blockIdx.y;
    const int qbase = (blockIdx.x * 4 + w) * 16;
    const long bh  = (long)(b * NHEAD + head);
    const bf16* Qp = Qb + bh * SEQ * HDIM;
    const bf16* Kp = Kb + bh * SEQ * HDIM;
    const bf16* Vp = Vt + bh * HDIM * SEQ;

    // stage K/V tile kb (32 keys) into LDS buffer p: 4 async b128 per lane
    auto stage = [&](int kb, int p) {
        // K: 32 rows of 128B; this wave covers rows w*8..w*8+7
        #pragma unroll
        for (int i = 0; i < 2; ++i) {
            int row = w * 8 + i * 4 + (lane >> 3);
            int seg = lane & 7;
            ASYNC_B128(Kp + (size_t)(kb + row) * HDIM + seg * 8,
                       &Klds[p][row * 64 + seg * 8]);
        }
        // V^T: 64 rows of 64B; this wave covers rows w*16..w*16+15
        #pragma unroll
        for (int i = 0; i < 2; ++i) {
            int row = w * 16 + i * 8 + (lane >> 2);
            int seg = lane & 3;
            ASYNC_B128(Vp + (size_t)row * SEQ + kb + seg * 8,
                       &Vlds[p][row * 32 + seg * 8]);
        }
    };

    // Q A-fragments, resident for the whole loop
    v16bf qf[2];
    {
        const bf16* qrow = Qp + (qbase + nn) * HDIM;
        #pragma unroll
        for (int cc = 0; cc < 2; ++cc)
            qf[cc] = cat16(*(const v8bf*)(qrow + cc * 32 + h * 8),
                           *(const v8bf*)(qrow + cc * 32 + 16 + h * 8));
    }

    float mrow[8], lrow[8];
    v8f acc[4];
    #pragma unroll
    for (int i = 0; i < 8; ++i) { mrow[i] = -3.0e38f; lrow[i] = 0.0f; }
    #pragma unroll
    for (int nt = 0; nt < 4; ++nt) acc[nt] = {};

    stage(0, 0);
    for (int kt = 0; kt < 64; ++kt) {
        const int p = kt & 1;
        WAIT_ASYNC();          // this wave's slice of buffer p is in LDS
        __syncthreads();       // every wave's slice is in LDS; prev reads done
        if (kt + 1 < 64) stage((kt + 1) * 32, 1 - p);   // overlap with compute

        // ---- S = Q K^T for two 16-key subtiles (from LDS)
        v8f s[2];
        #pragma unroll
        for (int sub = 0; sub < 2; ++sub) {
            const bf16* krow = &Klds[p][(sub * 16 + nn) * 64];
            v8f cf = {};
            #pragma unroll
            for (int cc = 0; cc < 2; ++cc) {
                v16bf bfr = cat16(*(const v8bf*)(krow + cc * 32 + h * 8),
                                  *(const v8bf*)(krow + cc * 32 + 16 + h * 8));
                cf = __builtin_amdgcn_wmma_f32_16x16x32_bf16(false, qf[cc], false, bfr,
                                                             (short)0, cf, false, false);
            }
            s[sub] = cf;
        }
        // ---- online softmax in exp2 domain (rows = 8*h+i)
        float tmax[8];
        #pragma unroll
        for (int i = 0; i < 8; ++i) {
            s[0][i] *= SOFTMAX_SCL; s[1][i] *= SOFTMAX_SCL;
            tmax[i] = fmaxf(s[0][i], s[1][i]);
        }
        #pragma unroll
        for (int off = 1; off < 16; off <<= 1) {
            #pragma unroll
            for (int i = 0; i < 8; ++i)
                tmax[i] = fmaxf(tmax[i], __shfl_xor(tmax[i], off, 32));
        }
        float corr[8], rsum[8];
        #pragma unroll
        for (int i = 0; i < 8; ++i) {
            float mn = fmaxf(mrow[i], tmax[i]);
            corr[i] = exp2f(mrow[i] - mn);
            mrow[i] = mn;
            s[0][i] = exp2f(s[0][i] - mn);
            s[1][i] = exp2f(s[1][i] - mn);
            rsum[i] = s[0][i] + s[1][i];
        }
        #pragma unroll
        for (int off = 1; off < 16; off <<= 1) {
            #pragma unroll
            for (int i = 0; i < 8; ++i)
                rsum[i] += __shfl_xor(rsum[i], off, 32);
        }
        #pragma unroll
        for (int i = 0; i < 8; ++i) {
            lrow[i] = lrow[i] * corr[i] + rsum[i];
            #pragma unroll
            for (int nt = 0; nt < 4; ++nt) acc[nt][i] *= corr[i];
        }
        // ---- P: C-layout -> A-layout via per-wave LDS (same-wave DS in-order)
        #pragma unroll
        for (int sub = 0; sub < 2; ++sub)
            #pragma unroll
            for (int i = 0; i < 8; ++i)
                Plds[w][8 * h + i][sub * 16 + nn] = (bf16)s[sub][i];
        const bf16* prow = &Plds[w][nn][0];
        v16bf pf = cat16(*(const v8bf*)(prow + h * 8),
                         *(const v8bf*)(prow + 16 + h * 8));
        // ---- O += P . V (V^T tile in LDS -> two 8-key runs per lane)
        #pragma unroll
        for (int nt = 0; nt < 4; ++nt) {
            const bf16* vrow = &Vlds[p][(nt * 16 + nn) * 32];
            v16bf vf = cat16(*(const v8bf*)(vrow + h * 8), *(const v8bf*)(vrow + 16 + h * 8));
            acc[nt] = __builtin_amdgcn_wmma_f32_16x16x32_bf16(false, pf, false, vf,
                                                              (short)0, acc[nt], false, false);
        }
    }
    // ---- finalize and write Y [B*T, 256] bf16
    #pragma unroll
    for (int i = 0; i < 8; ++i) {
        float inv = 1.0f / lrow[i];
        #pragma unroll
        for (int nt = 0; nt < 4; ++nt) acc[nt][i] *= inv;
    }
    #pragma unroll
    for (int i = 0; i < 8; ++i) {
        long row = (long)b * SEQ + qbase + 8 * h + i;
        bf16* yr = Y + row * HIDDEN + head * HDIM + nn;
        #pragma unroll
        for (int nt = 0; nt < 4; ++nt) yr[nt * 16] = (bf16)acc[nt][i];
    }
}

// ---------------------------------------------------------------------------
// Kernel 3: output projection.  out[m,n] = sum_k Y[m,k]*Wo[n,k] + bo[n] (f32)
// ---------------------------------------------------------------------------
__global__ void proj_kernel(const bf16* __restrict__ Y,
                            const bf16* __restrict__ Wo,
                            const float* __restrict__ bo,
                            float* __restrict__ out) {
    const int lane = threadIdx.x & 31;
    const int w    = threadIdx.x >> 5;
    const int h    = lane >> 4;
    const int m16  = lane & 15;
    const int ntile = blockIdx.x;             // 0..15
    const int mtile = blockIdx.y * 4 + w;     // 0..2047
    const long mbase = (long)mtile * 16;
    const int  nbase = ntile * 16;

    v8f c = {};
    #pragma unroll
    for (int kc = 0; kc < 8; ++kc) {
        const bf16* yrow = Y + (mbase + m16) * 256 + kc * 32;
        v16bf a = cat16(*(const v8bf*)(yrow + h * 8),
                        *(const v8bf*)(yrow + 16 + h * 8));
        const bf16* brow = Wo + (long)(nbase + m16) * 256 + kc * 32;
        v16bf b = cat16(*(const v8bf*)(brow + h * 8), *(const v8bf*)(brow + 16 + h * 8));
        c = __builtin_amdgcn_wmma_f32_16x16x32_bf16(false, a, false, b,
                                                    (short)0, c, false, false);
    }
    const float bias = bo[nbase + m16];
    #pragma unroll
    for (int i = 0; i < 8; ++i)
        out[(mbase + 8 * h + i) * 256 + nbase + m16] = c[i] + bias;
}

// ---------------------------------------------------------------------------
extern "C" void kernel_launch(void* const* d_in, const int* in_sizes, int n_in,
                              void* d_out, int out_size, void* d_ws, size_t ws_size,
                              hipStream_t stream) {
    const float* x    = (const float*)d_in[0];
    const float* Wqkv = (const float*)d_in[1];
    const float* bqkv = (const float*)d_in[2];
    const float* Wo   = (const float*)d_in[3];
    const float* bo   = (const float*)d_in[4];
    float* out = (float*)d_out;

    const size_t qn = (size_t)BATCH * NHEAD * SEQ * HDIM;   // 8,388,608 elems
    const size_t xn = (size_t)BATCH * SEQ * HIDDEN;         // 8,388,608 elems
    bf16* Qb    = (bf16*)d_ws;
    bf16* Kb    = Qb + qn;
    bf16* Vt    = Kb + qn;
    bf16* Yb    = Vt + qn;
    bf16* xb    = Yb + xn;
    bf16* Wqkvb = xb + xn;
    bf16* Wob   = Wqkvb + (size_t)3 * HIDDEN * HIDDEN;

    cvt_kernel<<<dim3((int)(xn / 4 / 256)), 256, 0, stream>>>(x, xb, (int)(xn / 4));
    cvt_kernel<<<dim3(192), 256, 0, stream>>>(Wqkv, Wqkvb, 3 * HIDDEN * HIDDEN / 4);
    cvt_kernel<<<dim3(64), 256, 0, stream>>>(Wo, Wob, HIDDEN * HIDDEN / 4);

    qkv_kernel<<<dim3(48, 512), 128, 0, stream>>>(xb, Wqkvb, bqkv, Qb, Kb, Vt);
    attn_kernel<<<dim3(32, NHEAD, BATCH), 128, 0, stream>>>(Qb, Kb, Vt, Yb);
    proj_kernel<<<dim3(16, 512), 128, 0, stream>>>(Yb, Wob, bo, out);
}
